// LocalTransformer_27547920236847
// MI455X (gfx1250) — matrix operand order repaired and checked by reading, not compiled, not run on hardware
//
#include <hip/hip_runtime.h>
#include <hip/hip_bf16.h>

// ---------------- problem constants ----------------
#define Bc      8
#define Nc      4096
#define Cc      128
#define Sc      1024
#define NSc     32
#define NHc     4
#define HDc     32
#define DOUTc   256
#define EPSc    1e-5f

typedef _Float16 v16h __attribute__((ext_vector_type(16)));
typedef _Float16 v8h  __attribute__((ext_vector_type(8)));
typedef float    v8f  __attribute__((ext_vector_type(8)));

// ---------------- WMMA helpers (gfx1250, wave32) ----------------
// A (16x32 f16): lane L: m=L&15, hf=L>>4; elems 0..15 -> K=k0+16*hf+i
// One 32B aligned/contiguous run -> vector load.
__device__ __forceinline__ v16h ldsA(const _Float16* X, int ld, int m0, int k0, int lane) {
  int m = lane & 15, hf = lane >> 4;
  const _Float16* p = X + (m0 + m) * ld + k0 + 16 * hf;
  return *(const v16h*)(p);
}
// B (32x16 f16), B[k][n] = W[(n0+n)*ld + k0 + k]: lane L: n=L&15, hf=L>>4; one 32B contiguous run
__device__ __forceinline__ v16h ldsB(const _Float16* W, int ld, int n0, int k0, int lane) {
  int n = lane & 15, hf = lane >> 4;
  const _Float16* p = W + (size_t)(n0 + n) * ld + k0 + 16 * hf;
  return *(const v16h*)(p);
}
__device__ __forceinline__ v8f wmma16(v16h a, v16h b, v8f c) {
  return __builtin_amdgcn_wmma_f32_16x16x32_f16(false, a, false, b, (short)0, c, false, false);
}

// ---------------- prep kernels ----------------
__global__ __launch_bounds__(256) void transpose_xyz_k(const float* __restrict__ xyz,
                                                       float* __restrict__ xyzT) {
  int i = blockIdx.x * 256 + threadIdx.x;              // B*N = 32768
  if (i >= Bc * Nc) return;
  int b = i / Nc, n = i % Nc;
#pragma unroll
  for (int d = 0; d < 3; ++d)
    xyzT[((size_t)b * Nc + n) * 3 + d] = xyz[((size_t)b * 3 + d) * Nc + n];
}

__global__ __launch_bounds__(256) void transpose_feat_k(const float* __restrict__ f,
                                                        float* __restrict__ fT) {
  int i = blockIdx.x * 256 + threadIdx.x;              // B*N*C = 4194304
  if (i >= Bc * Nc * Cc) return;
  int b = i / (Nc * Cc), r = i % (Nc * Cc);
  int n = r / Cc, c = r % Cc;
  fT[i] = f[((size_t)b * Cc + c) * Nc + n];
}

// f16 weight pack: [qkv 2*384*128][out 2*128*128][ff1 2*256*128][ff2 2*128*256]
#define WO_OUT 98304
#define WO_FF1 131072
#define WO_FF2 196608
#define WO_TOT 262144
__global__ __launch_bounds__(256) void cvt_w_k(const float* __restrict__ a, const float* __restrict__ b,
                                               const float* __restrict__ c, const float* __restrict__ d,
                                               _Float16* __restrict__ o) {
  int i = blockIdx.x * 256 + threadIdx.x;
  if (i >= WO_TOT) return;
  float v;
  if (i < WO_OUT)       v = a[i];
  else if (i < WO_FF1)  v = b[i - WO_OUT];
  else if (i < WO_FF2)  v = c[i - WO_FF1];
  else                  v = d[i - WO_FF2];
  o[i] = (_Float16)v;
}

// ---------------- FPS: one block per batch ----------------
__global__ __launch_bounds__(256) void fps_k(const float* __restrict__ xyzT,
                                             float* __restrict__ newxyz,     // [B,3,S]
                                             float* __restrict__ queries) {  // [B,S,3]
  __shared__ float dist[Nc];
  __shared__ float rv[256];
  __shared__ int   ri[256];
  __shared__ int   s_far;
  __shared__ float cc[3];
  int b = blockIdx.x, tid = threadIdx.x;
  const float* P = xyzT + (size_t)b * Nc * 3;
  for (int i = tid; i < Nc; i += 256) dist[i] = 1e10f;
  if (tid == 0) s_far = 0;
  __syncthreads();
  for (int it = 0; it < Sc; ++it) {
    if (tid == 0) {
      int far = s_far;
      float x = P[far * 3], y = P[far * 3 + 1], z = P[far * 3 + 2];
      cc[0] = x; cc[1] = y; cc[2] = z;
      newxyz[((size_t)b * 3 + 0) * Sc + it] = x;
      newxyz[((size_t)b * 3 + 1) * Sc + it] = y;
      newxyz[((size_t)b * 3 + 2) * Sc + it] = z;
      queries[((size_t)b * Sc + it) * 3 + 0] = x;
      queries[((size_t)b * Sc + it) * 3 + 1] = y;
      queries[((size_t)b * Sc + it) * 3 + 2] = z;
    }
    __syncthreads();
    float cx = cc[0], cy = cc[1], cz = cc[2];
    float best = -1.0f; int bi = Nc;
    for (int i = tid; i < Nc; i += 256) {
      float dx = P[i * 3] - cx, dy = P[i * 3 + 1] - cy, dz = P[i * 3 + 2] - cz;
      float dd = fminf(dist[i], dx * dx + dy * dy + dz * dz);
      dist[i] = dd;
      if (dd > best) { best = dd; bi = i; }        // strided i ascending -> first max kept
    }
    rv[tid] = best; ri[tid] = bi;
    __syncthreads();
    for (int s = 128; s > 0; s >>= 1) {
      if (tid < s) {
        float v2 = rv[tid + s]; int i2 = ri[tid + s];
        if (v2 > rv[tid] || (v2 == rv[tid] && i2 < ri[tid])) { rv[tid] = v2; ri[tid] = i2; }
      }
      __syncthreads();
    }
    if (tid == 0) s_far = ri[0];
    __syncthreads();
  }
}

// ---------------- KNN: 2 waves per block, one query per wave ----------------
__global__ __launch_bounds__(64) void knn_k(const float* __restrict__ xyzT,
                                            const float* __restrict__ queries,
                                            int* __restrict__ gidx) {
  __shared__ float d[2][Nc];                     // 32KB
  int wv = threadIdx.x >> 5, lane = threadIdx.x & 31;
  int g = blockIdx.x * 2 + wv;                   // b*S+s
  int b = g >> 10;
  float* dd = d[wv];
  const float* P = xyzT + (size_t)b * Nc * 3;
  float qx = queries[(size_t)g * 3], qy = queries[(size_t)g * 3 + 1], qz = queries[(size_t)g * 3 + 2];
  for (int i = lane; i < Nc; i += 32) {
    float dx = P[i * 3] - qx, dy = P[i * 3 + 1] - qy, dz = P[i * 3 + 2] - qz;
    dd[i] = dx * dx + dy * dy + dz * dz;
  }
  __syncthreads();
  for (int r = 0; r < NSc; ++r) {
    float best = 3.0e38f; int bi = Nc;
    for (int i = lane; i < Nc; i += 32) {
      float v = dd[i];
      if (v < best) { best = v; bi = i; }          // ascending i -> lowest index on ties
    }
#pragma unroll
    for (int m = 16; m > 0; m >>= 1) {
      float v2 = __shfl_xor(best, m, 32);
      int   i2 = __shfl_xor(bi, m, 32);
      if (v2 < best || (v2 == best && i2 < bi)) { best = v2; bi = i2; }
    }
    if (lane == 0) { gidx[(size_t)g * NSc + r] = bi; dd[bi] = 3.0e38f; }
    __syncthreads();
  }
}

// ---------------- fused group transformer ----------------
struct alignas(64) SMem {
  float x[NSc * Cc];                                   // 16KB activations f32
  union {
    float    ph[NSc * 64];                             // PE hidden (init phase)
    _Float16 qk[2 * NSc * Cc];                         // qh = qk[0..], kh = qk[4096..]
    _Float16 f1h[NSc * 2 * Cc];                        // FF hidden (relu, f16)
  } u1;                                                // 16KB
  union { _Float16 sc[NHc * NSc * NSc]; _Float16 ah[NSc * Cc]; } u2;   // 8KB
  union { _Float16 hn[NSc * Cc]; _Float16 sh[NHc * NSc * NSc]; float pooled[Cc]; } u3; // 8KB
  _Float16 vt[Cc * NSc];                               // V transposed [d][key], 8KB
  int   sidx[NSc];
  float gx[NSc * 3];
};

__device__ __forceinline__ void ln_row(const float* __restrict__ x, _Float16* __restrict__ o,
                                       const float* __restrict__ g, const float* __restrict__ b) {
  float m = 0.f;
  for (int c = 0; c < Cc; ++c) m += x[c];
  m *= (1.0f / Cc);
  float v = 0.f;
  for (int c = 0; c < Cc; ++c) { float t = x[c] - m; v += t * t; }
  v *= (1.0f / Cc);
  float rs = rsqrtf(v + EPSc);
  for (int c = 0; c < Cc; ++c) o[c] = (_Float16)((x[c] - m) * rs * g[c] + b[c]);
}

__global__ __launch_bounds__(128) void group_k(
    const float* __restrict__ xyzT, const float* __restrict__ featT, const int* __restrict__ gidx,
    const float* __restrict__ pe_w1, const float* __restrict__ pe_b1,
    const float* __restrict__ bn_g, const float* __restrict__ bn_b,
    const float* __restrict__ pe_w2, const float* __restrict__ pe_b2,
    const _Float16* __restrict__ wh,
    const float* __restrict__ in_b, const float* __restrict__ out_b,
    const float* __restrict__ ln1_g, const float* __restrict__ ln1_b,
    const float* __restrict__ ln2_g, const float* __restrict__ ln2_b,
    const float* __restrict__ ff1_b, const float* __restrict__ ff2_b,
    const float* __restrict__ fc_w, const float* __restrict__ fc_b,
    float* __restrict__ out2) {
  __shared__ SMem sm;
  int g = blockIdx.x;
  int b = g >> 10, s = g & 1023;
  int tid = threadIdx.x, lane = tid & 31, w = tid >> 5;

  // ---- gather indices / xyz ----
  if (tid < NSc) sm.sidx[tid] = gidx[(size_t)g * NSc + tid];
  __syncthreads();
  if (tid < NSc) {
    int idx = sm.sidx[tid];
    const float* p = xyzT + ((size_t)b * Nc + idx) * 3;
    sm.gx[tid * 3] = p[0]; sm.gx[tid * 3 + 1] = p[1]; sm.gx[tid * 3 + 2] = p[2];
  }
  __syncthreads();

  // ---- PE MLP hidden: h = relu(bn(gxyz @ w1^T + b1)) ----
  {
    int j = tid >> 2, k0 = (tid & 3) * 16;
    float gxv = sm.gx[j * 3], gyv = sm.gx[j * 3 + 1], gzv = sm.gx[j * 3 + 2];
    const float inv = rsqrtf(1.0f + EPSc);
    for (int k = k0; k < k0 + 16; ++k) {
      float h = gxv * pe_w1[k * 3] + gyv * pe_w1[k * 3 + 1] + gzv * pe_w1[k * 3 + 2] + pe_b1[k];
      h = h * inv * bn_g[k] + bn_b[k];
      sm.u1.ph[j * 64 + k] = fmaxf(h, 0.f);
    }
  }
  __syncthreads();

  // ---- x = gathered features + pe ----
  {
    int j = tid >> 2, c0 = (tid & 3) * 32;
    int idx = sm.sidx[j];
    const float* f = featT + ((size_t)b * Nc + idx) * Cc;
    const float* ph = &sm.u1.ph[j * 64];
    for (int c = c0; c < c0 + 32; ++c) {
      float acc = pe_b2[c];
      const float* w2 = pe_w2 + c * 64;
      for (int k = 0; k < 64; ++k) acc += ph[k] * w2[k];
      sm.x[j * Cc + c] = f[c] + acc;
    }
  }
  __syncthreads();

  const float scale = 0.17677669529663687f;  // 1/sqrt(32)
  _Float16* qh = sm.u1.qk;
  _Float16* kh = sm.u1.qk + NSc * Cc;

  for (int l = 0; l < 2; ++l) {
    const _Float16* Wqkv = wh + (size_t)l * 384 * 128;
    const _Float16* Wout = wh + WO_OUT + (size_t)l * 128 * 128;
    const _Float16* Wf1  = wh + WO_FF1 + (size_t)l * 256 * 128;
    const _Float16* Wf2  = wh + WO_FF2 + (size_t)l * 128 * 256;
    const float* inb = in_b + l * 384;
    const float* ob  = out_b + l * 128;
    const float* f1b = ff1_b + l * 256;
    const float* f2b = ff2_b + l * 128;

    // warm GL2 for this layer's weight stream (global_prefetch_b8)
    __builtin_prefetch(Wqkv + (size_t)tid * 384, 0, 0);
    __builtin_prefetch(Wout + (size_t)tid * 128, 0, 0);
    __builtin_prefetch(Wf1  + (size_t)tid * 256, 0, 0);
    __builtin_prefetch(Wf2  + (size_t)tid * 256, 0, 0);

    // LN1 -> hn (f16)
    if (tid < NSc) ln_row(sm.x + tid * Cc, sm.u3.hn + tid * Cc, ln1_g + l * 128, ln1_b + l * 128);
    __syncthreads();

    // ---- qkv = hn @ in_w^T + in_b : 2x24 tiles, K=128 ----
    for (int t = w; t < 48; t += 4) {
      int m0 = (t / 24) * 16, n0 = (t % 24) * 16;
      v8f acc = {};
#pragma unroll
      for (int kb = 0; kb < 4; ++kb)
        acc = wmma16(ldsA(sm.u3.hn, Cc, m0, kb * 32, lane), ldsB(Wqkv, 128, n0, kb * 32, lane), acc);
      int n = n0 + (lane & 15), hf8 = (lane >> 4) * 8;
      float bias = inb[n];
      if (n0 < 128) {               // uniform branch: whole tile is Q
#pragma unroll
        for (int j = 0; j < 8; ++j) qh[(m0 + j + hf8) * Cc + n] = (_Float16)(acc[j] + bias);
      } else if (n0 < 256) {        // whole tile is K
#pragma unroll
        for (int j = 0; j < 8; ++j) kh[(m0 + j + hf8) * Cc + (n - 128)] = (_Float16)(acc[j] + bias);
      } else {                      // whole tile is V -> transposed [d][key], packed 16B store
        v8h pk;
#pragma unroll
        for (int j = 0; j < 8; ++j) pk[j] = (_Float16)(acc[j] + bias);
        *(v8h*)(&sm.vt[(n - 256) * NSc + m0 + hf8]) = pk;
      }
    }
    __syncthreads();

    // ---- scores = (Q @ K^T) * scale : per head 2x2 tiles, K=32 ----
    for (int t = w; t < 16; t += 4) {
      int hh = t >> 2, m0 = ((t >> 1) & 1) * 16, n0 = (t & 1) * 16, k0 = hh * 32;
      v8f acc = {};
      acc = wmma16(ldsA(qh, Cc, m0, k0, lane), ldsB(kh, Cc, n0, k0, lane), acc);
      int n = n0 + (lane & 15), hf8 = (lane >> 4) * 8;
#pragma unroll
      for (int j = 0; j < 8; ++j)
        sm.u2.sc[hh * 1024 + (m0 + j + hf8) * NSc + n] = (_Float16)(acc[j] * scale);
    }
    __syncthreads();

    // ---- softmax over keys : 128 rows = 4 heads x 32 queries ----
    {
      const _Float16* row = &sm.u2.sc[(tid >> 5) * 1024 + (tid & 31) * NSc];
      float vv[NSc], mx = -3.0e38f;
#pragma unroll
      for (int k = 0; k < NSc; ++k) { vv[k] = (float)row[k]; mx = fmaxf(mx, vv[k]); }
      float sum = 0.f;
#pragma unroll
      for (int k = 0; k < NSc; ++k) { vv[k] = __expf(vv[k] - mx); sum += vv[k]; }
      float r = 1.0f / sum;
      _Float16* orow = &sm.u3.sh[(tid >> 5) * 1024 + (tid & 31) * NSc];
#pragma unroll
      for (int k = 0; k < NSc; ++k) orow[k] = (_Float16)(vv[k] * r);
    }
    __syncthreads();

    // ---- a = att @ V : per head 2x2 tiles, K=32 keys ----
    for (int t = w; t < 16; t += 4) {
      int hh = t >> 2, m0 = ((t >> 1) & 1) * 16;
      int d0 = hh * 32 + (t & 1) * 16;
      v8f acc = {};
      acc = wmma16(ldsA(sm.u3.sh + hh * 1024, NSc, m0, 0, lane), ldsB(sm.vt, NSc, d0, 0, lane), acc);
      int n = d0 + (lane & 15), hf8 = (lane >> 4) * 8;
#pragma unroll
      for (int j = 0; j < 8; ++j)
        sm.u2.ah[(m0 + j + hf8) * Cc + n] = (_Float16)acc[j];
    }
    __syncthreads();

    // ---- x += a @ out_w^T + out_b : 2x8 tiles, K=128 ----
    for (int t = w; t < 16; t += 4) {
      int m0 = (t >> 3) * 16, n0 = (t & 7) * 16;
      v8f acc = {};
#pragma unroll
      for (int kb = 0; kb < 4; ++kb)
        acc = wmma16(ldsA(sm.u2.ah, Cc, m0, kb * 32, lane), ldsB(Wout, 128, n0, kb * 32, lane), acc);
      int n = n0 + (lane & 15), hf8 = (lane >> 4) * 8;
      float bias = ob[n];
#pragma unroll
      for (int j = 0; j < 8; ++j)
        sm.x[(m0 + j + hf8) * Cc + n] += acc[j] + bias;
    }
    __syncthreads();

    // LN2 -> hn
    if (tid < NSc) ln_row(sm.x + tid * Cc, sm.u3.hn + tid * Cc, ln2_g + l * 128, ln2_b + l * 128);
    __syncthreads();

    // ---- f1 = relu(hn @ ff1^T + b1) : 2x16 tiles, K=128 ----
    for (int t = w; t < 32; t += 4) {
      int m0 = (t >> 4) * 16, n0 = (t & 15) * 16;
      v8f acc = {};
#pragma unroll
      for (int kb = 0; kb < 4; ++kb)
        acc = wmma16(ldsA(sm.u3.hn, Cc, m0, kb * 32, lane), ldsB(Wf1, 128, n0, kb * 32, lane), acc);
      int n = n0 + (lane & 15), hf8 = (lane >> 4) * 8;
      float bias = f1b[n];
#pragma unroll
      for (int j = 0; j < 8; ++j)
        sm.u1.f1h[(m0 + j + hf8) * 256 + n] = (_Float16)fmaxf(acc[j] + bias, 0.f);
    }
    __syncthreads();

    // ---- x += f1 @ ff2^T + b2 : 2x8 tiles, K=256 ----
    for (int t = w; t < 16; t += 4) {
      int m0 = (t >> 3) * 16, n0 = (t & 7) * 16;
      v8f acc = {};
#pragma unroll
      for (int kb = 0; kb < 8; ++kb)
        acc = wmma16(ldsA(sm.u1.f1h, 256, m0, kb * 32, lane), ldsB(Wf2, 256, n0, kb * 32, lane), acc);
      int n = n0 + (lane & 15), hf8 = (lane >> 4) * 8;
      float bias = f2b[n];
#pragma unroll
      for (int j = 0; j < 8; ++j)
        sm.x[(m0 + j + hf8) * Cc + n] += acc[j] + bias;
    }
    __syncthreads();
  }

  // ---- maxpool over neighbors + fc ----
  {
    float mx = sm.x[tid];
    for (int r = 1; r < NSc; ++r) mx = fmaxf(mx, sm.x[r * Cc + tid]);
    sm.u3.pooled[tid] = mx;
  }
  __syncthreads();
  for (int d = tid; d < DOUTc; d += 128) {
    float acc = fc_b[d];
    const float* wv = fc_w + d * Cc;
    for (int c = 0; c < Cc; ++c) acc += sm.u3.pooled[c] * wv[c];
    out2[((size_t)b * DOUTc + d) * Sc + s] = acc;
  }
}

// ---------------- host launch ----------------
extern "C" void kernel_launch(void* const* d_in, const int* in_sizes, int n_in,
                              void* d_out, int out_size, void* d_ws, size_t ws_size,
                              hipStream_t stream) {
  const float* xyz   = (const float*)d_in[0];
  const float* feat  = (const float*)d_in[1];
  const float* pe_w1 = (const float*)d_in[2];
  const float* pe_b1 = (const float*)d_in[3];
  const float* bn_g  = (const float*)d_in[4];
  const float* bn_b  = (const float*)d_in[5];
  const float* pe_w2 = (const float*)d_in[6];
  const float* pe_b2 = (const float*)d_in[7];
  const float* in_w  = (const float*)d_in[8];
  const float* in_b  = (const float*)d_in[9];
  const float* out_w = (const float*)d_in[10];
  const float* out_b = (const float*)d_in[11];
  const float* ln1_g = (const float*)d_in[12];
  const float* ln1_b = (const float*)d_in[13];
  const float* ln2_g = (const float*)d_in[14];
  const float* ln2_b = (const float*)d_in[15];
  const float* ff1_w = (const float*)d_in[16];
  const float* ff1_b = (const float*)d_in[17];
  const float* ff2_w = (const float*)d_in[18];
  const float* ff2_b = (const float*)d_in[19];
  const float* fc_w  = (const float*)d_in[20];
  const float* fc_b  = (const float*)d_in[21];

  char* wsb = (char*)d_ws;
  float*    xyzT    = (float*)(wsb + 0);          //  393216 B
  float*    featT   = (float*)(wsb + 393216);     // 16777216 B
  float*    queries = (float*)(wsb + 17170432);   //   98304 B
  int*      gidx    = (int*)  (wsb + 17268736);   // 1048576 B
  _Float16* wh      = (_Float16*)(wsb + 18317312);//  524288 B

  float* newxyz = (float*)d_out;                  // [B,3,S] = 24576
  float* out2   = (float*)d_out + Bc * 3 * Sc;    // [B,DOUT,S]

  transpose_xyz_k<<<dim3((Bc * Nc) / 256), dim3(256), 0, stream>>>(xyz, xyzT);
  transpose_feat_k<<<dim3((Bc * Nc * Cc) / 256), dim3(256), 0, stream>>>(feat, featT);
  cvt_w_k<<<dim3(WO_TOT / 256), dim3(256), 0, stream>>>(in_w, out_w, ff1_w, ff2_w, wh);
  fps_k<<<dim3(Bc), dim3(256), 0, stream>>>(xyzT, newxyz, queries);
  knn_k<<<dim3((Bc * Sc) / 2), dim3(64), 0, stream>>>(xyzT, queries, gidx);
  group_k<<<dim3(Bc * Sc), dim3(128), 0, stream>>>(
      xyzT, featT, gidx, pe_w1, pe_b1, bn_g, bn_b, pe_w2, pe_b2, wh,
      in_b, out_b, ln1_g, ln1_b, ln2_g, ln2_b, ff1_b, ff2_b, fc_w, fc_b, out2);
}
